// FactorizedEnsembleModel_51470888075968
// MI455X (gfx1250) — matrix operand order repaired and checked by reading, not compiled, not run on hardware
//
#include <hip/hip_runtime.h>
#include <hip/hip_bf16.h>

// ---------------- problem constants (from reference) ----------------
#define D_   18
#define E_   10
#define IN_  23
#define H_   128
#define B_   4096
#define BT   128            // batch rows per block
#define KP   136            // padded K stride (halfs) for H=128 arrays: 272B, 16B-aligned
#define MIN_LOGVAR (-10.0f)
#define MAX_LOGVAR (5.0f)

typedef __attribute__((ext_vector_type(16))) _Float16 v16h;
typedef __attribute__((ext_vector_type(8)))  float    v8f;

// Fast SiLU: v * sigmoid(v) with raw v_rcp_f32 (no IEEE div expansion)
__device__ __forceinline__ float silu_f(float v) {
    return v * __builtin_amdgcn_rcpf(1.0f + __expf(-v));
}
// Fast softplus via v_exp_f32 / v_log_f32 (only ~2 values per output row)
__device__ __forceinline__ float softplus_f(float v) {
    return (v > 20.0f) ? v : __logf(1.0f + __expf(v));
}

// A fragment (16x32 f16, M x K), contiguous K order used by this code (not AMD's table order):
//   lane L: M = L&15 ; halfs 0..15 hold K = (L<16?0:16)+0..15 (matches B fragment)
__device__ __forceinline__ v16h load_a_frag(const _Float16* base, int strideK, int lane, int ktile) {
    const int m  = lane & 15;
    const int kb = (lane >> 4) << 4;            // 0 or 16
    const _Float16* p = base + m * strideK + ktile;
    v16h a;
#pragma unroll
    for (int e = 0; e < 8; ++e) a[e]     = p[kb + e];
#pragma unroll
    for (int e = 0; e < 8; ++e) a[8 + e] = p[kb + 8 + e];
    return a;
}

// B fragment (32x16 f16, K x N) from a transposed [N][K] LDS image:
//   lane L: N = L&15 ; halfs 0..15 hold K = (L<16?0:16)+0..15   (contiguous in LDS)
__device__ __forceinline__ v16h load_b_frag(const _Float16* wT, int strideK, int lane, int ntile, int ktile) {
    const int n  = lane & 15;
    const int kb = (lane >> 4) << 4;            // 0 or 16
    const _Float16* p = wT + (ntile + n) * strideK + ktile + kb;
    v16h b;
#pragma unroll
    for (int e = 0; e < 16; ++e) b[e] = p[e];
    return b;
}

__global__ __launch_bounds__(256)
void ensemble_mlp_wmma(const float* __restrict__ x,   const float* __restrict__ masks,
                       const float* __restrict__ W1,  const float* __restrict__ b1,
                       const float* __restrict__ W2,  const float* __restrict__ b2,
                       const float* __restrict__ W3,  const float* __restrict__ b3,
                       float* __restrict__ out)
{
    // ~92 KB static LDS (gfx1250: 320 KB per WGP)
    __shared__ _Float16 ldsX  [BT * 32];     // masked x tile, K padded 23->32
    __shared__ _Float16 ldsW1T[H_ * 32];     // W1^T [n][k], K padded
    __shared__ _Float16 ldsW2T[H_ * KP];     // W2^T [n][k], padded stride
    __shared__ _Float16 ldsW3T[16 * KP];     // W3^T [n][k], N padded 2->16 with zeros
    __shared__ _Float16 ldsH  [BT * KP];     // h1 then h2 (wave-private 16-row strips)
    __shared__ float b1s[H_], b2s[H_], b3s[2];

    const int bid  = blockIdx.x;
    const int pair = bid >> 5;               // (d*E + e), 32 batch tiles per pair
    const int tile = bid & 31;
    const int d    = pair / E_;
    const int e    = pair % E_;
    const int b0   = tile * BT;

    const int tid     = threadIdx.x;
    const int lane    = tid & 31;            // wave32
    const int wave    = tid >> 5;            // 8 waves
    const int rowBase = wave * 16;           // each wave owns 16 batch rows

    const float* W1p = W1 + (size_t)pair * IN_ * H_;
    const float* W2p = W2 + (size_t)pair * H_ * H_;
    const float* W3p = W3 + (size_t)pair * H_ * 2;
    const float* mk  = masks + ((size_t)e * D_ + d) * IN_;   // m = transpose(masks,(1,0,2))

    // ---------------- cooperative stage: global f32 -> LDS f16 ----------------
    for (int idx = tid; idx < BT * 32; idx += 256) {          // x * mask, zero-padded K
        const int r = idx >> 5, k = idx & 31;
        const float v = (k < IN_) ? x[(size_t)(b0 + r) * IN_ + k] * mk[k] : 0.0f;
        ldsX[idx] = (_Float16)v;
    }
    for (int idx = tid; idx < H_ * 32; idx += 256) {          // W1^T, zero-padded K
        const int n = idx >> 5, k = idx & 31;
        const float v = (k < IN_) ? W1p[(size_t)k * H_ + n] : 0.0f;
        ldsW1T[idx] = (_Float16)v;
    }
    // W2^T: each thread converts a k-pair -> one packed dword LDS store.
    // Global reads stay n-coalesced across the wave for both k and k+1 streams.
    for (int idx = tid; idx < (H_ * H_) / 2; idx += 256) {
        const int n = idx & 127;
        const int k = (idx >> 7) << 1;
        union { _Float16 h[2]; unsigned u; } pk;
        pk.h[0] = (_Float16)W2p[(size_t)k       * H_ + n];
        pk.h[1] = (_Float16)W2p[(size_t)(k + 1) * H_ + n];
        *reinterpret_cast<unsigned*>(&ldsW2T[n * KP + k]) = pk.u;  // KP even, k even -> dword aligned
    }
    for (int idx = tid; idx < 16 * H_; idx += 256) {          // W3^T, N zero-padded 2->16
        const int n = idx >> 7, k = idx & 127;
        const float v = (n < 2) ? W3p[(size_t)k * 2 + n] : 0.0f;
        ldsW3T[n * KP + k] = (_Float16)v;
    }
    if (tid < H_) {
        b1s[tid] = b1[(size_t)pair * H_ + tid];
        b2s[tid] = b2[(size_t)pair * H_ + tid];
    }
    if (tid < 2) b3s[tid] = b3[(size_t)pair * 2 + tid];
    __syncthreads();

    // ---------------- layer 1: (16x32)·(32x128) per wave, 8 WMMAs ----------------
    {
        const v16h a = load_a_frag(ldsX + rowBase * 32, 32, lane, 0);
#pragma unroll
        for (int nt = 0; nt < 8; ++nt) {
            const float bv = b1s[nt * 16 + (lane & 15)];     // bias by output column
            v8f c;
#pragma unroll
            for (int r = 0; r < 8; ++r) c[r] = bv;
            const v16h b = load_b_frag(ldsW1T, 32, lane, nt * 16, 0);
            c = __builtin_amdgcn_wmma_f32_16x16x32_f16(false, a, false, b, (short)0, c, false, false);
            // SiLU, quantize to f16, store into wave-private strip of ldsH
#pragma unroll
            for (int r = 0; r < 8; ++r) {
                const int m = rowBase + r + ((lane >> 4) << 3);     // C layout: VGPR r -> row r (+8 upper half)
                ldsH[m * KP + nt * 16 + (lane & 15)] = (_Float16)silu_f(c[r]);
            }
        }
    }

    // ---------------- layer 2: (16x128)·(128x128) per wave, 32 WMMAs ----------------
    // All 8 N-tile accumulators kept in registers so h2 can overwrite h1 in LDS
    // (per-wave LDS ops are in-order, so the WAR on ldsH is safe).
    {
        v8f acc[8];
#pragma unroll
        for (int nt = 0; nt < 8; ++nt) {
            const float bv = b2s[nt * 16 + (lane & 15)];
#pragma unroll
            for (int r = 0; r < 8; ++r) acc[nt][r] = bv;
        }
        const _Float16* aBase = ldsH + rowBase * KP;
#pragma unroll
        for (int kt = 0; kt < 4; ++kt) {
            const v16h a = load_a_frag(aBase, KP, lane, kt * 32);
#pragma unroll
            for (int nt = 0; nt < 8; ++nt) {
                const v16h b = load_b_frag(ldsW2T, KP, lane, nt * 16, kt * 32);
                acc[nt] = __builtin_amdgcn_wmma_f32_16x16x32_f16(false, a, false, b, (short)0, acc[nt], false, false);
            }
        }
#pragma unroll
        for (int nt = 0; nt < 8; ++nt) {
#pragma unroll
            for (int r = 0; r < 8; ++r) {
                const int m = rowBase + r + ((lane >> 4) << 3);
                ldsH[m * KP + nt * 16 + (lane & 15)] = (_Float16)silu_f(acc[nt][r]);
            }
        }
    }
    // No block barrier needed: layer 3 reads only this wave's private ldsH strip,
    // and ldsW3T was staged before the first __syncthreads.

    // ---------------- layer 3: (16x128)·(128x16 zero-padded) per wave, 4 WMMAs ----------------
    {
        const int n = lane & 15;                 // output column; only 0/1 are real
        const float bv = (n < 2) ? b3s[n] : 0.0f;
        v8f c;
#pragma unroll
        for (int r = 0; r < 8; ++r) c[r] = bv;
        const _Float16* aBase = ldsH + rowBase * KP;
#pragma unroll
        for (int kt = 0; kt < 4; ++kt) {
            const v16h a = load_a_frag(aBase, KP, lane, kt * 32);
            const v16h b = load_b_frag(ldsW3T, KP, lane, 0, kt * 32);
            c = __builtin_amdgcn_wmma_f32_16x16x32_f16(false, a, false, b, (short)0, c, false, false);
        }
        // Only lanes holding column 0 (mean) / column 1 (logvar) store.
        if (n < 2) {
#pragma unroll
            for (int r = 0; r < 8; ++r) {
                const int m = rowBase + r + ((lane >> 4) << 3);
                const size_t gi = (size_t)pair * B_ + (size_t)(b0 + m);
                const float s = c[r];
                if (n == 0) {
                    out[gi] = s;                                        // mean, flat (D,E,B,1)
                } else {
                    float lv = MAX_LOGVAR - softplus_f(MAX_LOGVAR - s); // soft-clamp high
                    lv = MIN_LOGVAR + softplus_f(lv - MIN_LOGVAR);      // soft-clamp low
                    out[(size_t)D_ * E_ * B_ + gi] = lv;                // logvar after mean block
                }
            }
        }
    }
}

extern "C" void kernel_launch(void* const* d_in, const int* in_sizes, int n_in,
                              void* d_out, int out_size, void* d_ws, size_t ws_size,
                              hipStream_t stream) {
    const float* x     = (const float*)d_in[0];
    const float* masks = (const float*)d_in[1];
    const float* W1    = (const float*)d_in[2];
    const float* b1    = (const float*)d_in[3];
    const float* W2    = (const float*)d_in[4];
    const float* b2    = (const float*)d_in[5];
    const float* W3    = (const float*)d_in[6];
    const float* b3    = (const float*)d_in[7];
    float* out = (float*)d_out;

    const dim3 grid(D_ * E_ * (B_ / BT));    // 180 pairs x 32 batch tiles = 5760 blocks
    const dim3 block(256);                   // 8 wave32 waves
    hipLaunchKernelGGL(ensemble_mlp_wmma, grid, block, 0, stream,
                       x, masks, W1, b1, W2, b2, W3, b3, out);
}
